// DeformableAtt_55164559950045
// MI455X (gfx1250) — matrix operand order repaired and checked by reading, not compiled, not run on hardware
//
#include <hip/hip_runtime.h>

// ---------------------------------------------------------------------------
// Deformable attention, fused for MI455X (gfx1250, wave32, WMMA bf16).
// B=16, Q=1000, N=16, C=256, H=W=100, nh=8, d=32.
//
// Faithful index math: the torch-view scramble means query q2 (in batch b)
// attends to rows 16*q2..16*q2+15 of the [N*Q, C] sampled matrix, channel
// block h*32..h*32+31 per head => perfect 16-row WMMA tiles, K/V fully fused.
//
// Round-2 change: weights staged in LDS per workgroup via async-DMA
// (global_load_async_to_lds_b128 + s_wait_asynccnt), shared by all 8 waves.
// Padded LDS row stride (264 bf16) avoids 16-way bank conflicts on ds_load.
// ---------------------------------------------------------------------------

typedef unsigned short u16;
typedef unsigned int   u32;
typedef unsigned long long u64;
typedef __attribute__((ext_vector_type(16))) __bf16 v16bf;
typedef __attribute__((ext_vector_type(8)))  float  v8f;
typedef __attribute__((ext_vector_type(4)))  u32    u32x4;

union ABu { v16bf v; u32x4 u[2]; };   // 32 bytes: one WMMA bf16 A/B operand per lane

#define WROW 264                       // padded LDS row stride (bf16 elements)

__device__ __forceinline__ u16 f32_to_bf16(float f) {
  u32 x = __float_as_uint(f);
  x += 0x7FFFu + ((x >> 16) & 1u);    // round-to-nearest-even
  return (u16)(x >> 16);
}

// Async-DMA one 256x256 bf16 matrix (row-major) into LDS with padded rows.
// 8192 chunks of 16B; 256 threads x 32 iters; tracked by ASYNCcnt.
__device__ __forceinline__ void stage_weights_async(
    const u16* __restrict__ gsrc, u16* lds_base, int tid) {
#pragma unroll 4
  for (int i = 0; i < 32; ++i) {
    const int cidx = tid + i * 256;          // 16B chunk index
    const int row  = cidx >> 5;              // cidx / 32
    const int c16  = cidx & 31;
    const u16* g = gsrc + row * 256 + c16 * 8;
    const u32 loff = (u32)(uintptr_t)(lds_base + row * WROW + c16 * 8);
    const u64 ga = (u64)(uintptr_t)g;
    asm volatile("global_load_async_to_lds_b128 %0, %1, off"
                 :: "v"(loff), "v"(ga) : "memory");
  }
  asm volatile("s_wait_asynccnt 0" ::: "memory");
}

// --------------------------- kernel 1: fp32 -> bf16 ------------------------
__global__ __launch_bounds__(256) void cvt_bf16_kernel(
    const float* __restrict__ wq, const float* __restrict__ wk,
    const float* __restrict__ wv, const float* __restrict__ q,
    u16* __restrict__ o_wq, u16* __restrict__ o_wk,
    u16* __restrict__ o_wv, u16* __restrict__ o_q) {
  const int NW = 256 * 256;
  const int NQE = 16000 * 256;
  int i = blockIdx.x * blockDim.x + threadIdx.x;
  if (i < NW)            { o_wq[i]          = f32_to_bf16(wq[i]); }
  else if (i < 2 * NW)   { o_wk[i - NW]     = f32_to_bf16(wk[i - NW]); }
  else if (i < 3 * NW)   { o_wv[i - 2 * NW] = f32_to_bf16(wv[i - 2 * NW]); }
  else {
    int j = i - 3 * NW;
    if (j < NQE) o_q[j] = f32_to_bf16(q[j]);
  }
}

// ------------------- kernel 2: bilinear grid sample -> S_bf16 --------------
// S layout: [B][N*Q][C] with row rb = n*1000 + q  (matches torch [B,N,Q,C]).
__global__ __launch_bounds__(256) void sample_kernel(
    const float* __restrict__ x, const float* __restrict__ ref,
    u16* __restrict__ S) {
  const int blk = blockIdx.x;          // 0 .. 16*16000-1
  const int b  = blk / 16000;
  const int rb = blk % 16000;
  const int n  = rb / 1000;
  const int q  = rb % 1000;
  const int c  = threadIdx.x;          // channel, 256 threads

  const float rx = ref[(((size_t)(b * 1000 + q)) * 16 + n) * 2 + 0];
  const float ry = ref[(((size_t)(b * 1000 + q)) * 16 + n) * 2 + 1];
  const float gx = ((rx + 1.0f) * 100.0f - 1.0f) * 0.5f;  // align_corners=False
  const float gy = ((ry + 1.0f) * 100.0f - 1.0f) * 0.5f;
  const float x0f = floorf(gx), y0f = floorf(gy);
  const float fx = gx - x0f, fy = gy - y0f;
  const int x0 = (int)x0f, y0 = (int)y0f;

  const float* xb = x + (size_t)b * 256 * 10000 + (size_t)c * 10000;
  auto g = [&](int ix, int iy) -> float {
    if (ix < 0 || ix >= 100 || iy < 0 || iy >= 100) return 0.0f;  // zero pad
    return xb[iy * 100 + ix];
  };
  float v = g(x0,     y0    ) * (1.0f - fx) * (1.0f - fy)
          + g(x0 + 1, y0    ) * fx          * (1.0f - fy)
          + g(x0,     y0 + 1) * (1.0f - fx) * fy
          + g(x0 + 1, y0 + 1) * fx          * fy;
  S[((size_t)b * 16000 + rb) * 256 + c] = f32_to_bf16(v);
}

// ----------------- kernel 3: qp = q @ wq^T + bq  (WMMA bf16) ---------------
// grid.x = 1000 (M tiles of 16 rows); 8 waves, each wave owns 2 col-tiles.
__global__ __launch_bounds__(256) void qproj_kernel(
    const u16* __restrict__ Qb, const u16* __restrict__ Wq,
    const float* __restrict__ bq, float* __restrict__ qp) {
  const int w    = threadIdx.x >> 5;
  const int lane = threadIdx.x & 31;
  const int t    = blockIdx.x;
  const int M    = lane & 15;          // A row (lanes 0-15 / 16-31 both map M)
  const int hs   = lane >> 4;          // K half select

  // A operands: 16 rows x 256 cols of q_bf16, 8 k-steps, held in registers.
  const u16* arow = Qb + ((size_t)t * 16 + M) * 256;
  ABu A[8];
#pragma unroll
  for (int k = 0; k < 8; ++k) {
    const int c0 = k * 32 + hs * 16;
    A[k].u[0] = *(const u32x4*)(arow + c0);        // K = k*32+hs*16  .. +7
    A[k].u[1] = *(const u32x4*)(arow + c0 + 8);    // K = k*32+hs*16+8.. +7
  }

#pragma unroll
  for (int cti = 0; cti < 2; ++cti) {
    const int ct = w * 2 + cti;
    const float bias = bq[ct * 16 + M];
    v8f acc;
#pragma unroll
    for (int i = 0; i < 8; ++i) acc[i] = bias;
    const u16* wrow = Wq + (size_t)(ct * 16 + M) * 256;  // B col n = lane&15
#pragma unroll
    for (int k = 0; k < 8; ++k) {
      ABu Bop;
      const u16* p = wrow + k * 32 + hs * 16;      // 16 contiguous K values
      Bop.u[0] = *(const u32x4*)(p);
      Bop.u[1] = *(const u32x4*)(p + 8);
      acc = __builtin_amdgcn_wmma_f32_16x16x32_bf16(
          false, A[k].v, false, Bop.v, (short)0, acc, false, false);
    }
    // C layout: reg r -> row M_out = r + 8*hs, col N = lane&15
#pragma unroll
    for (int r = 0; r < 8; ++r)
      qp[((size_t)t * 16 + r + 8 * hs) * 256 + ct * 16 + M] = acc[r];
  }
}

// -------- kernel 4: fused K/V projection + softmax attention (WMMA) --------
// One wave per query. A tile (16 sample rows x 256 ch) loaded once, reused by
// both projections: 2 proj x 16 col-tiles x 8 k-steps = 256 WMMAs / query.
// Weights staged in (dynamic) LDS once per block, shared by all 8 waves.
#define WPB 8
__global__ __launch_bounds__(256) void fused_kv_attn_kernel(
    const u16* __restrict__ S,  const u16* __restrict__ Wk,
    const u16* __restrict__ Wv, const float* __restrict__ bk,
    const float* __restrict__ bv, const float* __restrict__ qp,
    float* __restrict__ out) {
  extern __shared__ __align__(16) u16 Wlds[];   // 256 x WROW bf16 = 135,168 B
  __shared__ float stage[WPB][16][16];   // 16x16 f32 col-tile stage per wave
  __shared__ float qp_lds[WPB][256];     // this query's qp row
  __shared__ float attn_lds[WPB][128];   // logits -> attn weights (8 heads x 16)

  const int tid  = threadIdx.x;
  const int w    = tid >> 5;
  const int lane = tid & 31;
  const int gq   = blockIdx.x * WPB + w;   // 0..15999 == b*1000+q2
  const int b    = gq / 1000;
  const int q2   = gq % 1000;
  const int M    = lane & 15;
  const int hs   = lane >> 4;
  const float scale = 0.17677669529663687f;  // 1/sqrt(32)

  // Async-DMA Wk into LDS (whole block cooperates).
  stage_weights_async(Wk, Wlds, tid);

  // Stage qp row; zero logits.
  const float* qprow = qp + (size_t)gq * 256;
  for (int j = lane; j < 256; j += 32) qp_lds[w][j] = qprow[j];
  for (int j = lane; j < 128; j += 32) attn_lds[w][j] = 0.0f;

  // A operands: S rows 16*q2+M of batch b (the 16 "keys" of this query).
  const u16* srow = S + ((size_t)b * 16000 + (size_t)q2 * 16 + M) * 256;
  ABu A[8];
#pragma unroll
  for (int k = 0; k < 8; ++k) {
    const int c0 = k * 32 + hs * 16;
    A[k].u[0] = *(const u32x4*)(srow + c0);
    A[k].u[1] = *(const u32x4*)(srow + c0 + 8);
  }
  __syncthreads();   // Wk staged, qp/attn ready

  // ---------------- K projection + logit accumulation ----------------
  for (int ct = 0; ct < 16; ++ct) {
    const float bias = bk[ct * 16 + M];
    v8f acc;
#pragma unroll
    for (int i = 0; i < 8; ++i) acc[i] = bias;
    const u16* wrow = Wlds + (ct * 16 + M) * WROW;   // B col n = lane&15 (LDS)
#pragma unroll
    for (int k = 0; k < 8; ++k) {
      ABu Bop;
      const u16* p = wrow + k * 32 + hs * 16;        // 32B contiguous, ds_load
      Bop.u[0] = *(const u32x4*)(p);
      Bop.u[1] = *(const u32x4*)(p + 8);
      acc = __builtin_amdgcn_wmma_f32_16x16x32_bf16(
          false, A[k].v, false, Bop.v, (short)0, acc, false, false);
    }
#pragma unroll
    for (int r = 0; r < 8; ++r) stage[w][r + 8 * hs][M] = acc[r];  // K[n2][c]
    asm volatile("s_wait_dscnt 0" ::: "memory");
    if (lane < 16) {             // lane = n2; dot 16 channels of this col-tile
      float p = 0.0f;
#pragma unroll
      for (int j = 0; j < 16; ++j) p += stage[w][lane][j] * qp_lds[w][ct * 16 + j];
      attn_lds[w][(ct >> 1) * 16 + lane] += p * scale;  // head = ct/2
    }
    asm volatile("s_wait_dscnt 0" ::: "memory");
  }

  // ---------------- softmax over n2 (16-lane head groups) ----------------
#pragma unroll
  for (int i = 0; i < 4; ++i) {  // 128 logits, 4 per lane; each 16-lane half = 1 head
    float v = attn_lds[w][i * 32 + lane];
    float m = v;
    for (int off = 8; off >= 1; off >>= 1) m = fmaxf(m, __shfl_xor(m, off, 16));
    float ex = __expf(v - m);
    float s = ex;
    for (int off = 8; off >= 1; off >>= 1) s += __shfl_xor(s, off, 16);
    attn_lds[w][i * 32 + lane] = ex / s;
  }

  // ---------------- swap LDS weights: Wk -> Wv ----------------
  __syncthreads();                       // all waves done reading Wk
  stage_weights_async(Wv, Wlds, tid);
  __syncthreads();                       // Wv staged

  // ---------------- V projection + attn-weighted reduction ----------------
  for (int ct = 0; ct < 16; ++ct) {
    const float bias = bv[ct * 16 + M];
    v8f acc;
#pragma unroll
    for (int i = 0; i < 8; ++i) acc[i] = bias;
    const u16* wrow = Wlds + (ct * 16 + M) * WROW;
#pragma unroll
    for (int k = 0; k < 8; ++k) {
      ABu Bop;
      const u16* p = wrow + k * 32 + hs * 16;
      Bop.u[0] = *(const u32x4*)(p);
      Bop.u[1] = *(const u32x4*)(p + 8);
      acc = __builtin_amdgcn_wmma_f32_16x16x32_bf16(
          false, A[k].v, false, Bop.v, (short)0, acc, false, false);
    }
#pragma unroll
    for (int r = 0; r < 8; ++r) stage[w][r + 8 * hs][M] = acc[r];  // V[n2][c]
    asm volatile("s_wait_dscnt 0" ::: "memory");
    if (lane < 16) {             // lane = local output channel
      const int h = ct >> 1;
      float o = 0.0f;
#pragma unroll
      for (int n2 = 0; n2 < 16; ++n2)
        o += attn_lds[w][h * 16 + n2] * stage[w][n2][lane];
      out[(size_t)gq * 256 + ct * 16 + lane] = o;
    }
    asm volatile("s_wait_dscnt 0" ::: "memory");
  }
}

// ------------------------------- launcher ----------------------------------
extern "C" void kernel_launch(void* const* d_in, const int* in_sizes, int n_in,
                              void* d_out, int out_size, void* d_ws, size_t ws_size,
                              hipStream_t stream) {
  const float* x   = (const float*)d_in[0];
  const float* q   = (const float*)d_in[1];
  const float* ref = (const float*)d_in[2];
  const float* wq  = (const float*)d_in[3];
  const float* bq  = (const float*)d_in[4];
  const float* wk  = (const float*)d_in[5];
  const float* bk  = (const float*)d_in[6];
  const float* wv  = (const float*)d_in[7];
  const float* bv  = (const float*)d_in[8];
  float* out = (float*)d_out;

  // workspace layout (bytes)
  char* ws = (char*)d_ws;
  u16*   WQb = (u16*)(ws + 0);          //  131,072  wq bf16
  u16*   WKb = (u16*)(ws + 131072);     //  131,072  wk bf16
  u16*   WVb = (u16*)(ws + 262144);     //  131,072  wv bf16
  u16*   Qb  = (u16*)(ws + 393216);     //  8,192,000  q bf16 [16000][256]
  float* QP  = (float*)(ws + 8585216);  // 16,384,000  qp f32 [16000][256]
  u16*   Sb  = (u16*)(ws + 24969216);   // 131,072,000 S bf16 [16][16000][256]

  const int ncvt = 3 * 65536 + 16000 * 256;
  cvt_bf16_kernel<<<(ncvt + 255) / 256, 256, 0, stream>>>(
      wq, wk, wv, q, WQb, WKb, WVb, Qb);
  sample_kernel<<<16 * 16000, 256, 0, stream>>>(x, ref, Sb);
  qproj_kernel<<<1000, 256, 0, stream>>>(Qb, WQb, bq, QP);

  const size_t wlds_bytes = (size_t)256 * WROW * sizeof(u16);  // 135,168 B dynamic LDS
  fused_kv_attn_kernel<<<16000 / WPB, 256, wlds_bytes, stream>>>(
      Sb, WKb, WVb, bk, bv, QP, out);
}
